// SFMStage_10977936408913
// MI455X (gfx1250) — compile-verified, matrix operand order checked by reading, not run on hardware
//
#include <hip/hip_runtime.h>

typedef _Float16 half_t;
typedef __attribute__((ext_vector_type(16))) _Float16 v16h;
typedef __attribute__((ext_vector_type(8)))  _Float16 v8h;
typedef __attribute__((ext_vector_type(8)))  float    v8f;

// Problem constants
constexpr int NROI = 512;
constexpr int BATCH = 2;
constexpr int C = 256;      // instance channels
constexpr int CS = 256;     // semantic channels
constexpr int HW = 200;
constexpr int OUTS = 14;    // 14x14
constexpr int P = 98;       // refined points
constexpr int PIX = OUTS * OUTS;          // 196
constexpr int FIN = C + CS + 3;           // 515
constexpr int FPAD = 544;                 // 515 padded to 17*32
constexpr int NSEM = BATCH * HW * HW;     // 80000
constexpr int NPTS = NROI * P;            // 50176
constexpr int NPIX = NROI * PIX;          // 100352

union Frag16 { v16h v; v8h h[2]; };

static __device__ __forceinline__ int clampi(int v, int lo, int hi) {
    return v < lo ? lo : (v > hi ? hi : v);
}

// ---------------------------------------------------------------------------
// TN GEMM with WMMA: Dt[n,m] = relu( sum_k A[m,k]*Bt[n,k] + bias[m] ).
// A row-major [M][K]; Bt row-major [N][K]; Dt row-major [N][M]. All fragment
// traffic is contiguous 16B (global_load_b128 / global_store_b128).
// One wave computes a 16(m) x 32(n) tile (A fragment reused by 2 WMMAs).
// Requires: K % 32 == 0, N % 32 == 0, M % 16 == 0, rows 16B-aligned.
// Fragment layouts per CDNA5 ISA 7.12.2 (wave32):
//   A 16x32 f16 : lane&15 = M row; lane>>4 picks K-halves {0-7,16-23}/{8-15,24-31}
//   B 32x16 f16 : lane&15 = N col; same K-half split
//   C/D 16x16 f32: lane&15 = N col; VGPR r -> M = r + 8*(lane>>4)
// ---------------------------------------------------------------------------
__global__ void gemm_tn_f16_relu_wmma(const half_t* __restrict__ A,
                                      const half_t* __restrict__ Bt,
                                      half_t* __restrict__ Dt,
                                      const float* __restrict__ bias,
                                      int M, int N, int K) {
    const int lane = threadIdx.x;
    const int mt   = blockIdx.x * blockDim.y + threadIdx.y;
    const int m0   = mt * 16;
    if (m0 >= M) return;
    const int n0   = blockIdx.y * 32;
    const int mloc = lane & 15;
    const int kh   = lane >> 4;

    const half_t* arow  = A  + (size_t)(m0 + mloc) * K + kh * 8;
    const half_t* brow0 = Bt + (size_t)(n0 + mloc) * K + kh * 8;
    const half_t* brow1 = brow0 + (size_t)16 * K;

    v8f acc0 = {0.f, 0.f, 0.f, 0.f, 0.f, 0.f, 0.f, 0.f};
    v8f acc1 = {0.f, 0.f, 0.f, 0.f, 0.f, 0.f, 0.f, 0.f};
    for (int k0 = 0; k0 < K; k0 += 32) {
        Frag16 a, b0, b1;
        a.h[0]  = *(const v8h*)(arow + k0);
        a.h[1]  = *(const v8h*)(arow + k0 + 16);
        b0.h[0] = *(const v8h*)(brow0 + k0);
        b0.h[1] = *(const v8h*)(brow0 + k0 + 16);
        b1.h[0] = *(const v8h*)(brow1 + k0);
        b1.h[1] = *(const v8h*)(brow1 + k0 + 16);
        __builtin_prefetch((const void*)(arow + k0 + 128), 0, 1);
        __builtin_prefetch((const void*)(brow0 + k0 + 128), 0, 1);
        __builtin_prefetch((const void*)(brow1 + k0 + 128), 0, 1);
        acc0 = __builtin_amdgcn_wmma_f32_16x16x32_f16(false, a.v, false, b0.v,
                                                      (short)0, acc0, false, false);
        acc1 = __builtin_amdgcn_wmma_f32_16x16x32_f16(false, a.v, false, b1.v,
                                                      (short)0, acc1, false, false);
    }
    const int mb = m0 + 8 * kh;
    v8h r0, r1;
#pragma unroll
    for (int r = 0; r < 8; ++r) {
        float b = bias[mb + r];
        r0[r] = (half_t)fmaxf(acc0[r] + b, 0.f);
        r1[r] = (half_t)fmaxf(acc1[r] + b, 0.f);
    }
    *(v8h*)(Dt + (size_t)(n0 + mloc) * M + mb)      = r0;
    *(v8h*)(Dt + (size_t)(n0 + 16 + mloc) * M + mb) = r1;
}

// ---------------------------------------------------------------------------
// 3x3 SAME conv on 14x14 tiles, implicit-GEMM WMMA, pixel-major layouts.
// IN : [roi][196][Cpad] f16 (zero-padded channels). Wt : [9][256][Cpad] f16.
// OUT: [(roi*196+p)][256] f16 = relu(conv + bias). Cpad % 32 == 0.
// One wave computes 32(o) x 16(p): B fragment shared by 2 WMMAs.
// Border pixels: unconditional b128 load + select (EXEC stays all-ones).
// ---------------------------------------------------------------------------
__global__ void conv3x3_relu_wmma(const half_t* __restrict__ IN,
                                  const half_t* __restrict__ Wt,
                                  half_t* __restrict__ OUTP,
                                  const float* __restrict__ bias,
                                  int Cpad) {
    const int lane = threadIdx.x;
    const int ot   = blockIdx.x * blockDim.y + threadIdx.y;   // 0..7
    const int o0   = ot * 32;
    const int p0   = blockIdx.y * 16;
    const int roi  = blockIdx.z;
    const int mloc = lane & 15;
    const int kh   = lane >> 4;
    const int pcol = p0 + mloc;
    const bool pv  = pcol < PIX;
    const int px = pcol % OUTS, py = pcol / OUTS;
    const half_t* inb = IN + (size_t)roi * PIX * Cpad;
    const int kchunks = Cpad >> 5;
    const v8h zero8 = {(half_t)0.f, (half_t)0.f, (half_t)0.f, (half_t)0.f,
                       (half_t)0.f, (half_t)0.f, (half_t)0.f, (half_t)0.f};

    v8f acc0 = {0.f, 0.f, 0.f, 0.f, 0.f, 0.f, 0.f, 0.f};
    v8f acc1 = {0.f, 0.f, 0.f, 0.f, 0.f, 0.f, 0.f, 0.f};
    for (int off = 0; off < 9; ++off) {
        const int dy = off / 3 - 1, dx = off % 3 - 1;
        const int yy = py + dy, xx = px + dx;
        const bool iv = pv && yy >= 0 && yy < OUTS && xx >= 0 && xx < OUTS;
        const ptrdiff_t sp = (ptrdiff_t)(yy * OUTS + xx);
        const half_t* brow  = inb + sp * Cpad + kh * 8;
        const half_t* wrow0 = Wt + ((size_t)off * 256 + o0 + mloc) * Cpad + kh * 8;
        const half_t* wrow1 = wrow0 + (size_t)16 * Cpad;
        for (int kc = 0; kc < kchunks; ++kc) {
            const int kb = kc * 32;
            Frag16 a0, a1, b;
            a0.h[0] = *(const v8h*)(wrow0 + kb);
            a0.h[1] = *(const v8h*)(wrow0 + kb + 16);
            a1.h[0] = *(const v8h*)(wrow1 + kb);
            a1.h[1] = *(const v8h*)(wrow1 + kb + 16);
            v8h bl0 = *(const v8h*)(brow + kb);
            v8h bl1 = *(const v8h*)(brow + kb + 16);
            b.h[0] = iv ? bl0 : zero8;
            b.h[1] = iv ? bl1 : zero8;
            acc0 = __builtin_amdgcn_wmma_f32_16x16x32_f16(false, a0.v, false, b.v,
                                                          (short)0, acc0, false, false);
            acc1 = __builtin_amdgcn_wmma_f32_16x16x32_f16(false, a1.v, false, b.v,
                                                          (short)0, acc1, false, false);
        }
    }
    if (pv) {
        const int mb0 = o0 + 8 * kh;
        v8h r0, r1;
#pragma unroll
        for (int r = 0; r < 8; ++r) {
            r0[r] = (half_t)fmaxf(acc0[r] + bias[mb0 + r], 0.f);
            r1[r] = (half_t)fmaxf(acc1[r] + bias[mb0 + 16 + r], 0.f);
        }
        half_t* orow = OUTP + (size_t)(roi * PIX + pcol) * 256;
        *(v8h*)(orow + mb0)      = r0;
        *(v8h*)(orow + mb0 + 16) = r1;
    }
}

// --------------------------- small helper kernels ---------------------------
__global__ void k_f32_to_f16(const float* __restrict__ s, half_t* __restrict__ d, int n) {
    int t = blockIdx.x * blockDim.x + threadIdx.x;
    if (t < n) d[t] = (half_t)s[t];
}

// semantic_feat [B][CS][40000] f32 -> [pix = b*40000+p][CS] f16 (Bt layout)
__global__ void k_pack_sem(const float* __restrict__ src, half_t* __restrict__ dst) {
    int t = blockIdx.x * blockDim.x + threadIdx.x;
    if (t >= NSEM * CS) return;
    int pix = t / CS;
    int c = t - pix * CS;
    int b = pix / (HW * HW);
    int p = pix - b * HW * HW;
    dst[t] = (half_t)src[((size_t)b * CS + c) * (HW * HW) + p];
}

// conv weights [O][Cin][3][3] f32 -> [9][O][Cpad] f16, zero-padded channels.
__global__ void k_pack_conv_w(const float* __restrict__ src, half_t* __restrict__ dst,
                              int Cin, int Cpad) {
    int t = blockIdx.x * blockDim.x + threadIdx.x;
    if (t >= 9 * 256 * Cpad) return;
    int off = t / (256 * Cpad);
    int r = t - off * 256 * Cpad;
    int o = r / Cpad;
    int ci = r - o * Cpad;
    dst[t] = (ci < Cin) ? (half_t)src[((size_t)o * Cin + ci) * 9 + off] : (half_t)0.f;
}

// Per-ROI label-selected 1x1 convs: inst/det predictions (f32, exact).
__global__ void k_inst_det(const float* __restrict__ feats, const int* __restrict__ labels,
                           const float* __restrict__ w_inst, const float* __restrict__ b_inst,
                           const float* __restrict__ w_det, const float* __restrict__ b_det,
                           float* __restrict__ instp, float* __restrict__ detp) {
    int t = blockIdx.x * blockDim.x + threadIdx.x;
    if (t >= NPIX) return;
    int roi = t / PIX, p = t - roi * PIX;
    int lab = labels[roi];
    const float* f  = feats + (size_t)roi * C * PIX + p;
    const float* wi = w_inst + lab * C;
    const float* wd = w_det + lab * C;
    float ai = 0.f, ad = 0.f;
    for (int c = 0; c < C; ++c) {
        float v = f[(size_t)c * PIX];
        ai += wi[c] * v;
        ad += wd[c] * v;
    }
    instp[t] = ai + b_inst[lab];
    detp[t]  = ad + b_det[lab];
}

// Exact top-98 of 196 by smallest |det| (rank == sorted position, ties by index).
__global__ void k_topk(const float* __restrict__ detp, int* __restrict__ pidx,
                       int* __restrict__ inv) {
    __shared__ float sa[PIX];
    int roi = blockIdx.x, tid = threadIdx.x;
    if (tid < PIX) sa[tid] = fabsf(detp[roi * PIX + tid]);
    __syncthreads();
    if (tid < PIX) {
        float mine = sa[tid];
        int rank = 0;
        for (int j = 0; j < PIX; ++j) {
            float aj = sa[j];
            rank += (aj < mine) || (aj == mine && j < tid);
        }
        int s = -1;
        if (rank < P) { pidx[roi * P + rank] = tid; s = rank; }
        inv[roi * PIX + tid] = s;
    }
}

struct BiLin { int o00, o01, o10, o11; float w00, w01, w10, w11; };
static __device__ __forceinline__ BiLin bilin200(float u, float v, int b) {
    float fx = floorf(u), fy = floorf(v);
    float wx = u - fx, wy = v - fy;
    int x0 = clampi((int)fx, 0, HW - 1);
    int x1 = x0 + 1 > HW - 1 ? HW - 1 : x0 + 1;
    int y0 = clampi((int)fy, 0, HW - 1);
    int y1 = y0 + 1 > HW - 1 ? HW - 1 : y0 + 1;
    BiLin r;
    int base = b * HW * HW;
    r.o00 = base + y0 * HW + x0; r.o01 = base + y0 * HW + x1;
    r.o10 = base + y1 * HW + x0; r.o11 = base + y1 * HW + x1;
    r.w00 = (1.f - wx) * (1.f - wy); r.w01 = wx * (1.f - wy);
    r.w10 = (1.f - wx) * wy;         r.w11 = wx * wy;
    return r;
}

// Vectorized 4-corner blend of one 256-channel row block (8 channels).
static __device__ __forceinline__ v8h blend8(const half_t* r00, const half_t* r01,
                                             const half_t* r10, const half_t* r11,
                                             int c8, const BiLin& g) {
    v8h a = *(const v8h*)(r00 + c8 * 8);
    v8h b = *(const v8h*)(r01 + c8 * 8);
    v8h c = *(const v8h*)(r10 + c8 * 8);
    v8h d = *(const v8h*)(r11 + c8 * 8);
    v8h res;
#pragma unroll
    for (int j = 0; j < 8; ++j)
        res[j] = (half_t)(g.w00 * (float)a[j] + g.w01 * (float)b[j] +
                          g.w10 * (float)c[j] + g.w11 * (float)d[j]);
    return res;
}

// H0 [t = roi*98+pp][512] f16: ch 0..255 fine (sem gather), 256..511 coarse.
__global__ void k_build_h0(const half_t* __restrict__ sem, const int* __restrict__ pidx,
                           const float* __restrict__ rois, const float* __restrict__ feats,
                           half_t* __restrict__ H0) {
    int t = blockIdx.x * blockDim.x + threadIdx.x;
    if (t >= NPTS) return;
    int roi = t / P;
    int idx = pidx[t];
    int b = (int)rois[roi * 5 + 0];
    float x1 = rois[roi * 5 + 1], y1 = rois[roi * 5 + 2];
    float x2 = rois[roi * 5 + 3], y2 = rois[roi * 5 + 4];
    float rx = ((idx % OUTS) + 0.5f) / OUTS;
    float ry = ((idx / OUTS) + 0.5f) / OUTS;
    float u = (x1 + rx * (x2 - x1)) * 0.25f - 0.5f;
    float v = (y1 + ry * (y2 - y1)) * 0.25f - 0.5f;
    BiLin g = bilin200(u, v, b);
    const half_t* r00 = sem + (size_t)g.o00 * CS;
    const half_t* r01 = sem + (size_t)g.o01 * CS;
    const half_t* r10 = sem + (size_t)g.o10 * CS;
    const half_t* r11 = sem + (size_t)g.o11 * CS;
    half_t* hr = H0 + (size_t)t * (C + CS);
    for (int c8 = 0; c8 < CS / 8; ++c8)
        *(v8h*)(hr + c8 * 8) = blend8(r00, r01, r10, r11, c8, g);
    for (int c = 0; c < C; ++c)
        hr[CS + c] = (half_t)feats[((size_t)roi * C + c) * PIX + idx];
}

// FUSED [(roi*196+p)][FPAD] f16: refined | ins_sem | instp | detp | mask | 0-pad.
__global__ void k_build_fused(const half_t* __restrict__ sem, const float* __restrict__ sempred,
                              const half_t* __restrict__ H3, const int* __restrict__ inv,
                              const float* __restrict__ feats, const float* __restrict__ instp,
                              const float* __restrict__ detp, const float* __restrict__ rois,
                              half_t* __restrict__ fused, float* __restrict__ mask) {
    int t = blockIdx.x * blockDim.x + threadIdx.x;
    if (t >= NPIX) return;
    int roi = t / PIX, p = t - roi * PIX;
    int x = p % OUTS, y = p / OUTS;
    int b = (int)rois[roi * 5 + 0];
    float x1 = rois[roi * 5 + 1], y1 = rois[roi * 5 + 2];
    float x2 = rois[roi * 5 + 3], y2 = rois[roi * 5 + 4];
    float jx = (x + 0.5f) / OUTS, jy = (y + 0.5f) / OUTS;
    float u = (x1 + jx * (x2 - x1)) * 0.25f - 0.5f;
    float v = (y1 + jy * (y2 - y1)) * 0.25f - 0.5f;
    BiLin g = bilin200(u, v, b);
    half_t* fb = fused + (size_t)t * FPAD;
    int slot = inv[t];
    if (slot >= 0) {
        const half_t* hr = H3 + (size_t)(roi * P + slot) * C;
        for (int c8 = 0; c8 < C / 8; ++c8)
            *(v8h*)(fb + c8 * 8) = *(const v8h*)(hr + c8 * 8);
    } else {
        for (int c = 0; c < C; ++c)
            fb[c] = (half_t)feats[((size_t)roi * C + c) * PIX + p];
    }
    const half_t* r00 = sem + (size_t)g.o00 * CS;
    const half_t* r01 = sem + (size_t)g.o01 * CS;
    const half_t* r10 = sem + (size_t)g.o10 * CS;
    const half_t* r11 = sem + (size_t)g.o11 * CS;
    for (int c8 = 0; c8 < CS / 8; ++c8)
        *(v8h*)(fb + C + c8 * 8) = blend8(r00, r01, r10, r11, c8, g);
    fb[C + CS]     = (half_t)instp[t];
    fb[C + CS + 1] = (half_t)detp[t];
    float m = g.w00 * sempred[g.o00] + g.w01 * sempred[g.o01] +
              g.w10 * sempred[g.o10] + g.w11 * sempred[g.o11];
    mask[t] = m;
    fb[C + CS + 2] = (half_t)m;
    for (int c = FIN; c < FPAD; ++c) fb[c] = (half_t)0.f;
}

__global__ void k_copy_preds(const float* __restrict__ instp, const float* __restrict__ detp,
                             float* __restrict__ out) {
    int t = blockIdx.x * blockDim.x + threadIdx.x;
    if (t >= 2 * NPIX) return;
    out[t] = (t < NPIX) ? instp[t] : detp[t - NPIX];
}

// Final 28x28 assembly: half-pixel bilinear for 'up' (ch<256), align-corners
// for inst/det/mask (ch 256..258). C3 is pixel-major [(roi*196+p)][256].
__global__ void k_finalize(const half_t* __restrict__ C3, const float* __restrict__ instp,
                           const float* __restrict__ detp, const float* __restrict__ mask,
                           float* __restrict__ out) {
    const size_t TOT = (size_t)NROI * 259 * 784;
    size_t t = (size_t)blockIdx.x * blockDim.x + threadIdx.x;
    if (t >= TOT) return;
    int roi = (int)(t / 203056);
    int r = (int)(t - (size_t)roi * 203056);
    int ch = r / 784;
    int q = r - ch * 784;
    int oy = q / 28, ox = q - oy * 28;
    float val;
    if (ch < C) {
        float sy = oy * 0.5f - 0.25f, sx = ox * 0.5f - 0.25f;
        int iy = (int)floorf(sy); float wy = sy - iy;
        int ix = (int)floorf(sx); float wx = sx - ix;
        int y0 = clampi(iy, 0, OUTS - 1), y1 = clampi(iy + 1, 0, OUTS - 1);
        int x0 = clampi(ix, 0, OUTS - 1), x1 = clampi(ix + 1, 0, OUTS - 1);
        const half_t* s = C3 + (size_t)roi * PIX * 256 + ch;
        float v00 = (float)s[(y0 * OUTS + x0) * 256];
        float v01 = (float)s[(y0 * OUTS + x1) * 256];
        float v10 = (float)s[(y1 * OUTS + x0) * 256];
        float v11 = (float)s[(y1 * OUTS + x1) * 256];
        val = v00 * (1.f - wx) * (1.f - wy) + v01 * wx * (1.f - wy) +
              v10 * (1.f - wx) * wy + v11 * wx * wy;
        val = fmaxf(val, 0.f);
    } else {
        const float* s = (ch == C ? instp : (ch == C + 1 ? detp : mask)) + roi * PIX;
        float sy = oy * (13.f / 27.f), sx = ox * (13.f / 27.f);
        int y0 = (int)sy; if (y0 > OUTS - 2) y0 = OUTS - 2;
        float wy = sy - y0; int y1 = y0 + 1;
        int x0 = (int)sx; if (x0 > OUTS - 2) x0 = OUTS - 2;
        float wx = sx - x0; int x1 = x0 + 1;
        float r0 = s[y0 * OUTS + x0] * (1.f - wy) + s[y1 * OUTS + x0] * wy;
        float r1 = s[y0 * OUTS + x1] * (1.f - wy) + s[y1 * OUTS + x1] * wy;
        val = r0 * (1.f - wx) + r1 * wx;
    }
    out[2 * NPIX + t] = val;
}

// ---------------------------------------------------------------------------
extern "C" void kernel_launch(void* const* d_in, const int* in_sizes, int n_in,
                              void* d_out, int out_size, void* d_ws, size_t ws_size,
                              hipStream_t stream) {
    const float* feats    = (const float*)d_in[0];
    const float* semfeat  = (const float*)d_in[1];
    const float* sempred  = (const float*)d_in[2];
    const float* rois     = (const float*)d_in[3];
    const int*   labels   = (const int*)d_in[4];
    const float* w_sem_in = (const float*)d_in[5];
    const float* b_sem_in = (const float*)d_in[6];
    const float* w_inst   = (const float*)d_in[7];
    const float* b_inst   = (const float*)d_in[8];
    const float* w_det    = (const float*)d_in[9];
    const float* b_det    = (const float*)d_in[10];
    const float* w_fc0    = (const float*)d_in[11];
    const float* b_fc0    = (const float*)d_in[12];
    const float* w_fc1    = (const float*)d_in[13];
    const float* b_fc1    = (const float*)d_in[14];
    const float* w_fc2    = (const float*)d_in[15];
    const float* b_fc2    = (const float*)d_in[16];
    const float* w_fuse0  = (const float*)d_in[17];
    const float* b_fuse0  = (const float*)d_in[18];
    const float* w_fuse1  = (const float*)d_in[19];
    const float* b_fuse1  = (const float*)d_in[20];
    const float* w_out    = (const float*)d_in[21];
    const float* b_out    = (const float*)d_in[22];
    float* out = (float*)d_out;

    // ---- workspace layout; total ~275 MB ----
    char* ws = (char*)d_ws;
    size_t o = 0;
    auto alloc = [&](size_t bytes) { size_t r = o; o += (bytes + 255) & ~(size_t)255; return r; };
    const size_t O_SEMIN = alloc((size_t)NSEM * CS * 2);          // Bt input (reused: C2 w/ O_SEM)
    const size_t O_SEM   = alloc((size_t)NSEM * CS * 2);          // relu(sem), pixel-major
    const size_t O_WSEM  = alloc((size_t)CS * CS * 2);
    const size_t O_WFC0  = alloc((size_t)C * (C + CS) * 2);
    const size_t O_WFC1  = alloc((size_t)C * C * 2);
    const size_t O_WFC2  = alloc((size_t)C * C * 2);
    const size_t O_WOUT  = alloc((size_t)C * C * 2);
    const size_t O_WF0   = alloc((size_t)9 * 256 * FPAD * 2);     // repacked, padded
    const size_t O_WF1   = alloc((size_t)9 * 256 * 256 * 2);
    const size_t O_INSTP = alloc((size_t)NPIX * 4);
    const size_t O_DETP  = alloc((size_t)NPIX * 4);
    const size_t O_PIDX  = alloc((size_t)NPTS * 4);
    const size_t O_INV   = alloc((size_t)NPIX * 4);
    const size_t O_MASK  = alloc((size_t)NPIX * 4);
    const size_t O_H0    = alloc((size_t)NPTS * (C + CS) * 2);    // reused: H2, C1
    const size_t O_H1    = alloc((size_t)NPTS * C * 2);           // reused: H3
    const size_t O_FUSED = alloc((size_t)NPIX * FPAD * 2);        // reused: C3
    alloc(65536);                                                  // guard for conv halo reads
    (void)ws_size;

    half_t* semin  = (half_t*)(ws + O_SEMIN);
    half_t* sem    = (half_t*)(ws + O_SEM);
    half_t* wsem16 = (half_t*)(ws + O_WSEM);
    half_t* wfc016 = (half_t*)(ws + O_WFC0);
    half_t* wfc116 = (half_t*)(ws + O_WFC1);
    half_t* wfc216 = (half_t*)(ws + O_WFC2);
    half_t* wout16 = (half_t*)(ws + O_WOUT);
    half_t* wf016  = (half_t*)(ws + O_WF0);
    half_t* wf116  = (half_t*)(ws + O_WF1);
    float* instp = (float*)(ws + O_INSTP);
    float* detp  = (float*)(ws + O_DETP);
    int*   pidx  = (int*)(ws + O_PIDX);
    int*   inv   = (int*)(ws + O_INV);
    float* maskb = (float*)(ws + O_MASK);
    half_t* H0   = (half_t*)(ws + O_H0);
    half_t* H1   = (half_t*)(ws + O_H1);
    half_t* H2   = (half_t*)(ws + O_H0);     // reuse (H0 dead after fc0)
    half_t* H3   = (half_t*)(ws + O_H1);     // reuse (H1 dead after fc1)
    half_t* FUSED = (half_t*)(ws + O_FUSED);
    half_t* C1   = (half_t*)(ws + O_H0);     // reuse (h buffers dead)
    half_t* C2   = (half_t*)(ws + O_SEMIN);  // reuse (sem buffers dead)
    half_t* C3   = (half_t*)(ws + O_FUSED);  // reuse (fused dead after conv0)

    auto cvt = [&](const float* s, half_t* d, int n) {
        k_f32_to_f16<<<(n + 255) / 256, 256, 0, stream>>>(s, d, n);
    };
    // 1. weight conversions / repacks
    cvt(w_sem_in, wsem16, CS * CS);
    cvt(w_fc0, wfc016, C * (C + CS));
    cvt(w_fc1, wfc116, C * C);
    cvt(w_fc2, wfc216, C * C);
    cvt(w_out, wout16, C * C);
    k_pack_conv_w<<<(9 * 256 * FPAD + 255) / 256, 256, 0, stream>>>(w_fuse0, wf016, FIN, FPAD);
    k_pack_conv_w<<<(9 * 256 * 256 + 255) / 256, 256, 0, stream>>>(w_fuse1, wf116, 256, 256);

    // 2. pack semantic features into Bt layout [80000][256] f16
    {
        int n = NSEM * CS;
        k_pack_sem<<<(n + 255) / 256, 256, 0, stream>>>(semfeat, semin);
    }
    // 3. sem = relu(1x1 conv): TN GEMM [256x256] x [80000x256]^T -> [80000][256]
    gemm_tn_f16_relu_wmma<<<dim3(4, NSEM / 32), dim3(32, 4), 0, stream>>>(
        wsem16, semin, sem, b_sem_in, CS, NSEM, CS);
    // 4. label-selected instance/detail predictions (f32)
    k_inst_det<<<(NPIX + 255) / 256, 256, 0, stream>>>(
        feats, labels, w_inst, b_inst, w_det, b_det, instp, detp);
    // 5. top-98 uncertain points per ROI
    k_topk<<<NROI, 256, 0, stream>>>(detp, pidx, inv);
    // 6. point feature matrix H0 [50176][512]
    k_build_h0<<<(NPTS + 255) / 256, 256, 0, stream>>>(sem, pidx, rois, feats, H0);
    // 7-9. fc0/fc1/fc2 TN GEMMs (N = 50176 = 1568*32)
    gemm_tn_f16_relu_wmma<<<dim3(4, NPTS / 32), dim3(32, 4), 0, stream>>>(
        wfc016, H0, H1, b_fc0, C, NPTS, C + CS);
    gemm_tn_f16_relu_wmma<<<dim3(4, NPTS / 32), dim3(32, 4), 0, stream>>>(
        wfc116, H1, H2, b_fc1, C, NPTS, C);
    gemm_tn_f16_relu_wmma<<<dim3(4, NPTS / 32), dim3(32, 4), 0, stream>>>(
        wfc216, H2, H3, b_fc2, C, NPTS, C);
    // 10. fused tensor [100352][544] (scatter refined + gathers + pred maps)
    k_build_fused<<<(NPIX + 255) / 256, 256, 0, stream>>>(
        sem, sempred, H3, inv, feats, instp, detp, rois, FUSED, maskb);
    // 11. fuse conv0: Cin 515(pad 544) -> C1 [100352][256]
    conv3x3_relu_wmma<<<dim3(2, 13, NROI), dim3(32, 4), 0, stream>>>(
        FUSED, wf016, C1, b_fuse0, FPAD);
    // 12. fuse conv1: Cin 256 -> C2 [100352][256]
    conv3x3_relu_wmma<<<dim3(2, 13, NROI), dim3(32, 4), 0, stream>>>(
        C1, wf116, C2, b_fuse1, 256);
    // 13. w_out 1x1: TN GEMM -> C3 [100352][256]
    gemm_tn_f16_relu_wmma<<<dim3(4, NPIX / 32), dim3(32, 4), 0, stream>>>(
        wout16, C2, C3, b_out, C, NPIX, C);
    // 14. first two outputs
    k_copy_preds<<<(2 * NPIX + 255) / 256, 256, 0, stream>>>(instp, detp, out);
    // 15. final 28x28 upsample + concat (417 MB of writes; HBM-bound)
    {
        size_t tot = (size_t)NROI * 259 * 784;
        int blocks = (int)((tot + 255) / 256);
        k_finalize<<<blocks, 256, 0, stream>>>(C3, instp, detp, maskb, out);
    }
}
